// DistanceLoss_20890720927816
// MI455X (gfx1250) — matrix operand order repaired and checked by reading, not compiled, not run on hardware
//
#include <hip/hip_runtime.h>
#include <hip/hip_fp16.h>

// ---------------- problem constants ----------------
#define NQ      256
#define NS      25
#define SEQ     16
#define DIN     2048
#define HDIM    1024
#define TT      120          // C(16,2)
#define WAY     5
#define MQ_ROWS (NQ*SEQ)     // 4096
#define MS_ROWS (NS*SEQ)     // 400
#define MROWS   (MQ_ROWS+MS_ROWS)   // 4496 (=281*16)
#define QE_ROWS (NQ*TT)      // 30720 (=1920*16)
#define SE_ROWS (NS*TT)      // 3000
#define K1      2048
#define N1      2048
#define K2      1024
#define BIGF    3.4e37f

typedef __attribute__((ext_vector_type(16))) _Float16 v16h;
typedef __attribute__((ext_vector_type(8)))  float    v8f;

union V16 { uint4 q[2]; v16h v; };

// ---------------- 1a. convert X (queries ++ support) to fp16 ----------------
__global__ void cvt_x_kernel(const float* __restrict__ q, const float* __restrict__ s,
                             __half* __restrict__ xcat) {
    int idx = blockIdx.x * blockDim.x + threadIdx.x;      // 4496*2048 elements
    int r = idx >> 11, k = idx & 2047;
    float v = (r < MQ_ROWS) ? q[idx] : s[(size_t)(r - MQ_ROWS) * DIN + k];
    xcat[idx] = __float2half(v);
}

// ---------------- 1b. reorganize W: Wt[n][k], n<1024 -> W[n][k], else W[n-1024][2048+k]
__global__ void cvt_w_kernel(const float* __restrict__ W, __half* __restrict__ Wt) {
    int idx = blockIdx.x * blockDim.x + threadIdx.x;      // 2048*2048 elements
    int n = idx >> 11, k = idx & 2047;
    float v = W[(size_t)(n & (HDIM - 1)) * (2 * DIN) + ((n >> 10) ? DIN : 0) + k];
    Wt[idx] = __float2half(v);
}

// ---------------- 2. GEMM1: Act[4496][2048] = Xcat * Wt^T  (WMMA f16->f32) ----
// block = 256 (8 waves). Block covers M16 x N128. A tile (16x2048 f16 = 64KB) in LDS.
__global__ __launch_bounds__(256)
void gemm1_kernel(const __half* __restrict__ X, const __half* __restrict__ Wt,
                  float* __restrict__ Act) {
    extern __shared__ char smem[];
    __half* As = (__half*)smem;                    // 16*2048 halves
    const int tid  = threadIdx.x;
    const int lane = tid & 31, wave = tid >> 5;
    const int l16  = lane & 15;
    const int half8 = (lane < 16) ? 0 : 8;
    const int mt = blockIdx.x >> 4;                // 0..280
    const int nb = blockIdx.x & 15;
    const int mbase = mt * 16;
    const int nbase = nb * 128 + wave * 16;

    // cooperative A stage: 16 rows * 2048 halves = 4096 uint4
    {
        const uint4* src = (const uint4*)(X + (size_t)mbase * K1);
        uint4* dst = (uint4*)As;
        for (int i = tid; i < 4096; i += 256) dst[i] = src[i];
    }
    __syncthreads();

    v8f acc = {0.f,0.f,0.f,0.f,0.f,0.f,0.f,0.f};
    const int dA = (lane < 16) ? 0 : 8;
    const int dB = (lane < 16) ? 0 : 8;
    #pragma unroll 8
    for (int kt = 0; kt < K1 / 32; ++kt) {
        const int kb = kt * 32;
        V16 a, b;
        const uint4* ap = (const uint4*)(As + l16 * K1 + kb + dA);
        a.q[0] = ap[0]; a.q[1] = ap[2];            // K and K+16
        const uint4* bp = (const uint4*)(Wt + (size_t)(nbase + l16) * K1 + kb + dB);
        b.q[0] = bp[0]; b.q[1] = bp[2];            // K and K+16
        acc = __builtin_amdgcn_wmma_f32_16x16x32_f16(false, a.v, false, b.v,
                                                     (short)0, acc, false, false);
    }
    #pragma unroll
    for (int i = 0; i < 8; ++i)
        Act[(size_t)(mbase + half8 + i) * N1 + nbase + l16] = acc[i];
}

// ---------------- 3. combine pair projections + bias + relu -> f16 emb + norms
__global__ __launch_bounds__(256)
void embed_kernel(const float* __restrict__ Act, const float* __restrict__ bias,
                  __half* __restrict__ qe, __half* __restrict__ se,
                  float* __restrict__ q2, float* __restrict__ s2) {
    __shared__ float red[256];
    const int r = blockIdx.x;                      // 0..33719
    const int tid = threadIdx.x;
    int samp, t; __half* outp; float* normp; int rowoff;
    if (r < QE_ROWS) { samp = r / TT; t = r % TT; rowoff = samp * SEQ;
                       outp = qe + (size_t)r * HDIM; normp = q2 + r; }
    else { int rs = r - QE_ROWS; samp = rs / TT; t = rs % TT; rowoff = MQ_ROWS + samp * SEQ;
           outp = se + (size_t)rs * HDIM; normp = s2 + rs; }
    // decode tuple t -> (i,j), itertools.combinations order
    int i = 0, tt = t;
    while (true) { int cnt = (SEQ - 1) - i; if (tt < cnt) break; tt -= cnt; ++i; }
    const int j = i + 1 + tt;
    const int h0 = tid * 4;
    const float4 va = *(const float4*)(Act + (size_t)(rowoff + i) * N1 + h0);
    const float4 vb = *(const float4*)(Act + (size_t)(rowoff + j) * N1 + HDIM + h0);
    const float4 bb = *(const float4*)(bias + h0);
    __half h[4]; float nrm = 0.f;
    const float e0 = fmaxf(va.x + vb.x + bb.x, 0.f);
    const float e1 = fmaxf(va.y + vb.y + bb.y, 0.f);
    const float e2 = fmaxf(va.z + vb.z + bb.z, 0.f);
    const float e3 = fmaxf(va.w + vb.w + bb.w, 0.f);
    h[0] = __float2half(e0); h[1] = __float2half(e1);
    h[2] = __float2half(e2); h[3] = __float2half(e3);
    #pragma unroll
    for (int x = 0; x < 4; ++x) { float f = __half2float(h[x]); nrm += f * f; }
    *(uint2*)(outp + h0) = *(uint2*)h;
    red[tid] = nrm; __syncthreads();
    for (int sft = 128; sft > 0; sft >>= 1) {
        if (tid < sft) red[tid] += red[tid + sft];
        __syncthreads();
    }
    if (tid == 0) *normp = red[0];
}

// ---- async stage of one 16x1024-f16 B tile into an LDS buffer (ASYNCcnt) ----
__device__ __forceinline__ void stage_B_async(const __half* se, __half* buf,
                                              int nbase, int tid) {
    const unsigned long long sbase = (unsigned long long)se;
    #pragma unroll
    for (int it = 0; it < 8; ++it) {
        const int idx = tid + it * 256;            // 0..2047 uint4 chunks
        const int col = idx >> 7, off = idx & 127;
        int srow = nbase + col; if (srow >= SE_ROWS) srow = SE_ROWS - 1;
        const unsigned ldsoff = (unsigned)(size_t)(buf + (size_t)idx * 8);
        const unsigned goff = (unsigned)(((unsigned)srow * K2 + (unsigned)off * 8u) * 2u);
        asm volatile("global_load_async_to_lds_b128 %0, %1, %2"
                     :: "v"(ldsoff), "v"(goff), "s"(sbase) : "memory");
    }
}
__device__ __forceinline__ void wait_async0() {
    asm volatile("s_wait_asynccnt 0x0" ::: "memory");
}

// ---------------- 4. distance GEMM + fused sqrt + per-class min ----------------
// block = 256 (8 waves); wave w owns M-tile (blockIdx*8+w); all waves sweep N,
// sharing each 16x1024 f16 B tile via double-buffered LDS filled with
// GLOBAL_LOAD_ASYNC_TO_LDS_B128 (overlapped with WMMA on the other buffer).
// LDS: 2*32KB (B ping-pong) + 8*32KB (A per wave) = 320KB.
__global__ __launch_bounds__(256)
void gemm2_min_kernel(const __half* __restrict__ qe, const __half* __restrict__ se,
                      const float* __restrict__ q2, const float* __restrict__ s2,
                      const int* __restrict__ labels, float* __restrict__ minOut) {
    extern __shared__ char smem[];
    __half* Bs0 = (__half*)smem;                       // 16*1024 halves
    __half* Bs1 = (__half*)(smem + 16 * K2 * 2);
    __half* As  = (__half*)(smem + 2 * 16 * K2 * 2);   // 8 waves * 16*1024
    const int tid  = threadIdx.x;
    const int lane = tid & 31, wave = tid >> 5;
    const int l16  = lane & 15;
    const int half8 = (lane < 16) ? 0 : 8;
    const int mbase = (blockIdx.x * 8 + wave) * 16;
    __half* Aw = As + (size_t)wave * 16 * K2;

    // per-wave A stage: 2048 uint4 (regular path; read only by this wave, in-order LDS)
    {
        const uint4* src = (const uint4*)(qe + (size_t)mbase * K2);
        uint4* dst = (uint4*)Aw;
        for (int i = lane; i < 2048; i += 32) dst[i] = src[i];
    }
    float q2r[8];
    #pragma unroll
    for (int i = 0; i < 8; ++i) q2r[i] = q2[mbase + half8 + i];

    float minv[8][5];
    #pragma unroll
    for (int i = 0; i < 8; ++i)
        #pragma unroll
        for (int c = 0; c < WAY; ++c) minv[i][c] = BIGF;

    const int dA = (lane < 16) ? 0 : 8;
    const int dB = (lane < 16) ? 0 : 8;
    const int NTILES = (SE_ROWS + 15) / 16;            // 188

    // prologue: stage tile 0
    stage_B_async(se, Bs0, 0, tid);
    wait_async0();
    __syncthreads();

    for (int nt = 0; nt < NTILES; ++nt) {
        const int nbase = nt * 16;
        // kick off async stage of the NEXT tile into the other buffer
        if (nt + 1 < NTILES)
            stage_B_async(se, (nt & 1) ? Bs0 : Bs1, (nt + 1) * 16, tid);
        const __half* Bcur = (nt & 1) ? Bs1 : Bs0;

        v8f acc = {0.f,0.f,0.f,0.f,0.f,0.f,0.f,0.f};
        #pragma unroll
        for (int kt = 0; kt < K2 / 32; ++kt) {
            const int kb = kt * 32;
            V16 a, b;
            const uint4* ap = (const uint4*)(Aw + l16 * K2 + kb + dA);
            a.q[0] = ap[0]; a.q[1] = ap[2];
            const uint4* bp = (const uint4*)(Bcur + l16 * K2 + kb + dB);
            b.q[0] = bp[0]; b.q[1] = bp[2];
            acc = __builtin_amdgcn_wmma_f32_16x16x32_f16(false, a.v, false, b.v,
                                                         (short)0, acc, false, false);
        }
        // fused epilogue: dist = sqrt(max(q2+s2-2*dot, eps)); per-class running min
        const int col = nbase + l16;
        const bool valid = col < SE_ROWS;
        const int colc = valid ? col : SE_ROWS - 1;
        const float s2c = s2[colc];
        const int cls = labels[colc / TT];
        #pragma unroll
        for (int i = 0; i < 8; ++i) {
            float d = q2r[i] + s2c - 2.0f * acc[i];
            d = sqrtf(fmaxf(d, 1e-12f));
            if (!valid) d = BIGF;
            #pragma unroll
            for (int c = 0; c < WAY; ++c) {
                float cand = (cls == c) ? d : BIGF;
                minv[i][c] = fminf(minv[i][c], cand);
            }
        }
        // next buffer fully landed for every wave before anyone reads it
        wait_async0();
        __syncthreads();
    }

    // reduce mins across the 16 lanes of each half (N direction)
    #pragma unroll
    for (int i = 0; i < 8; ++i)
        #pragma unroll
        for (int c = 0; c < WAY; ++c) {
            float v = minv[i][c];
            v = fminf(v, __shfl_xor(v, 1, 16));
            v = fminf(v, __shfl_xor(v, 2, 16));
            v = fminf(v, __shfl_xor(v, 4, 16));
            v = fminf(v, __shfl_xor(v, 8, 16));
            if (l16 == 0) minOut[(size_t)(mbase + half8 + i) * WAY + c] = v;
        }
}

// ---------------- 5. logits[n][c] = -mean_t min ----------------
__global__ __launch_bounds__(128)
void logits_kernel(const float* __restrict__ minOut, float* __restrict__ out) {
    __shared__ float red[128];
    const int n = blockIdx.x, tid = threadIdx.x;
    #pragma unroll
    for (int c = 0; c < WAY; ++c) {
        float v = (tid < TT) ? minOut[(size_t)(n * TT + tid) * WAY + c] : 0.f;
        red[tid] = v; __syncthreads();
        for (int sft = 64; sft > 0; sft >>= 1) {
            if (tid < sft) red[tid] += red[tid + sft];
            __syncthreads();
        }
        if (tid == 0) out[n * WAY + c] = -red[0] / (float)TT;
        __syncthreads();
    }
}

// ---------------- launch ----------------
extern "C" void kernel_launch(void* const* d_in, const int* in_sizes, int n_in,
                              void* d_out, int out_size, void* d_ws, size_t ws_size,
                              hipStream_t stream) {
    const float* support = (const float*)d_in[0];
    const float* queries = (const float*)d_in[1];
    const int*   labels  = (const int*)d_in[2];
    const float* W       = (const float*)d_in[3];
    const float* bias    = (const float*)d_in[4];
    float* out = (float*)d_out;

    char* ws = (char*)d_ws;
    size_t o = 0;
    auto alloc = [&](size_t bytes) { size_t p = o; o += (bytes + 255) & ~(size_t)255; return p; };
    __half* Xcat   = (__half*)(ws + alloc((size_t)MROWS * K1 * 2));
    __half* Wt     = (__half*)(ws + alloc((size_t)N1 * K1 * 2));
    float*  Act    = (float*) (ws + alloc((size_t)MROWS * N1 * 4));
    __half* qe     = (__half*)(ws + alloc((size_t)QE_ROWS * K2 * 2));
    __half* se     = (__half*)(ws + alloc((size_t)SE_ROWS * K2 * 2));
    float*  q2     = (float*) (ws + alloc((size_t)QE_ROWS * 4));
    float*  s2     = (float*) (ws + alloc((size_t)SE_ROWS * 4));
    float*  minOut = (float*) (ws + alloc((size_t)QE_ROWS * WAY * 4));
    (void)ws_size; (void)in_sizes; (void)n_in; (void)out_size;

    cvt_x_kernel<<<(MROWS * K1) / 256, 256, 0, stream>>>(queries, support, Xcat);
    cvt_w_kernel<<<(N1 * K1) / 256, 256, 0, stream>>>(W, Wt);
    gemm1_kernel<<<(MROWS / 16) * 16, 256, 16 * K1 * 2, stream>>>(Xcat, Wt, Act);
    embed_kernel<<<QE_ROWS + SE_ROWS, 256, 0, stream>>>(Act, bias, qe, se, q2, s2);
    gemm2_min_kernel<<<QE_ROWS / (16 * 8), 256, (16 * K2 * 2) * 10, stream>>>(
        qe, se, q2, s2, labels, minOut);
    logits_kernel<<<NQ, 128, 0, stream>>>(minOut, out);
}
